// ExpertGroup_11665131176299
// MI455X (gfx1250) — matrix operand order repaired and checked by reading, not compiled, not run on hardware
//
#include <hip/hip_runtime.h>
#include <hip/hip_bf16.h>

// ---------------------------------------------------------------------------
// MoE adapter block on MI455X (gfx1250, wave32, WMMA).
//   1) cvt f32->f16 for x + all GEMM weights (workspace)
//   2) shared   = silu(x @ up_w^T)              WMMA f16, f32 accum
//   3) pre      = shared @ adapt_w^T            WMMA
//   4) per-token winning expert (last i with w>0): adapter AxA + LayerNorm f32
//   5) t        = h_ln @ expert_proj^T          WMMA
//   6) out      = shared + 0.1 * mask * (t @ output_proj^T)   WMMA fused epi
// GEMMs use LDS double buffering with GLOBAL_LOAD_ASYNC_TO_LDS_B128 (ASYNCcnt)
// when available, so tile k+1 streams into LDS while tile k feeds the WMMAs.
// ---------------------------------------------------------------------------

typedef __attribute__((ext_vector_type(16))) _Float16 v16h;
typedef __attribute__((ext_vector_type(8)))  float    v8f;

union Frag16 { v16h v; unsigned int u[8]; };

#if __has_builtin(__builtin_amdgcn_global_load_async_to_lds_b128)
#define HAVE_ASYNC_LDS 1
// the builtin takes pointers to GCC-style <4 x i32> vectors in AS(1)/AS(3)
typedef int b128_t __attribute__((vector_size(16)));
typedef __attribute__((address_space(1))) b128_t* gb128_p;
typedef __attribute__((address_space(3))) b128_t* lb128_p;
#endif

__device__ __forceinline__ void copy16_to_lds(const void* gsrc, void* lds_dst) {
#ifdef HAVE_ASYNC_LDS
    __builtin_amdgcn_global_load_async_to_lds_b128(
        (gb128_p)(void*)gsrc, (lb128_p)lds_dst, 0, 0);
#else
    *reinterpret_cast<uint4*>(lds_dst) =
        *reinterpret_cast<const uint4*>(gsrc);
#endif
}

__device__ __forceinline__ void wait_async_stage() {
#ifdef HAVE_ASYNC_LDS
#if __has_builtin(__builtin_amdgcn_s_wait_asynccnt)
    __builtin_amdgcn_s_wait_asynccnt(0);
#else
    asm volatile("s_wait_asynccnt 0x0" ::: "memory");
#endif
#endif
}

__global__ void cvt_f32_to_f16(const float* __restrict__ src,
                               _Float16* __restrict__ dst, int n) {
    int i = blockIdx.x * blockDim.x + threadIdx.x;
    int stride = gridDim.x * blockDim.x;
    for (; i < n; i += stride) dst[i] = (_Float16)src[i];
}

// ---------------------------------------------------------------------------
// Tiled WMMA GEMM:  C[M,N] = A[M,K] * B[N,K]^T   (both operands f16, row-major)
// Block: 256 threads = 8 waves (2 x 4), block tile 64 x 256, wave tile 32 x 64.
// LDS double-buffered; async staging overlaps HBM with WMMA.
// EPI: 0 = store f16 to outH
//      1 = silu -> store f32 to outF and f16 to outH
//      2 = outF = baseF + (sel[m] >= 0 ? 0.1f * acc : 0)
// ---------------------------------------------------------------------------
template <int EPI>
__global__ __launch_bounds__(256)
void gemm_wmma(const _Float16* __restrict__ Ah, const _Float16* __restrict__ Bh,
               int M, int N, int K,
               float* __restrict__ outF, _Float16* __restrict__ outH,
               const float* __restrict__ baseF, const int* __restrict__ sel) {
    constexpr int KT   = 32;
    constexpr int LDR  = KT + 8;       // 80-byte rows keep b128 LDS ops aligned
    constexpr int ABUF = 64 * LDR;     // halves per A stage
    constexpr int BBUF = 256 * LDR;    // halves per B stage
    __shared__ _Float16 As[2 * ABUF];
    __shared__ _Float16 Bs[2 * BBUF];

    const int t      = threadIdx.x;
    const int lane   = t & 31;
    const int wave   = t >> 5;
    const int waveM  = (wave >> 2) * 32;     // 0 / 32
    const int waveN  = (wave & 3) * 64;      // 0..192
    const int laneHi = lane >> 4;            // 0 / 1
    const int lane16 = lane & 15;

    const int blockM = blockIdx.y * 64;
    const int blockN = blockIdx.x * 256;

    // staging geometry: each thread moves one 16B chunk of A, four of B
    const int srow = t >> 2;                 // 0..63
    const int sc8  = (t & 3) * 8;            // 0,8,16,24

    v8f acc[2][4];
#pragma unroll
    for (int i = 0; i < 2; ++i)
#pragma unroll
        for (int j = 0; j < 4; ++j)
#pragma unroll
            for (int e = 0; e < 8; ++e) acc[i][j][e] = 0.0f;

    auto stage = [&](int buf, int k0) {
        _Float16* Ad = As + buf * ABUF;
        _Float16* Bd = Bs + buf * BBUF;
        copy16_to_lds(Ah + (size_t)(blockM + srow) * K + k0 + sc8,
                      &Ad[srow * LDR + sc8]);
#pragma unroll
        for (int q = 0; q < 4; ++q) {
            const int brow = srow + q * 64;  // 0..255
            const int n    = blockN + brow;
            if (n < N) {
                copy16_to_lds(Bh + (size_t)n * K + k0 + sc8,
                              &Bd[brow * LDR + sc8]);
            } else {
                uint4 z;
                z.x = z.y = z.z = z.w = 0u;
                *reinterpret_cast<uint4*>(&Bd[brow * LDR + sc8]) = z;
            }
        }
    };

    const int nSteps = K / KT;               // all K here are multiples of 32
    stage(0, 0);

    for (int kb = 0; kb < nSteps; ++kb) {
        const int cur = kb & 1;
        // make buffer `cur` visible to every wave
        wait_async_stage();
        __syncthreads();
        // kick off the next stage while we compute on `cur`
        if (kb + 1 < nSteps) stage(cur ^ 1, (kb + 1) * KT);

        const _Float16* Ac = As + cur * ABUF;
        const _Float16* Bc = Bs + cur * BBUF;

        // --- pack fragments per CDNA5 16-bit WMMA VGPR layouts -------------
        Frag16 a[2], b[4];
#pragma unroll
        for (int i = 0; i < 2; ++i) {
            const _Float16* ap = &Ac[(waveM + i * 16 + lane16) * LDR];
#pragma unroll
            for (int v = 0; v < 8; ++v) {
                // A 16x32: lane<16: K {0..7,16..23}; lane>=16: {8..15,24..31}
                const int kk = 2 * (v & 3) + laneHi * 8 + (v >> 2) * 16;
                a[i].u[v] = *reinterpret_cast<const unsigned int*>(ap + kk);
            }
        }
#pragma unroll
        for (int j = 0; j < 4; ++j) {
            const _Float16* bp = &Bc[(waveN + j * 16 + lane16) * LDR];
#pragma unroll
            for (int v = 0; v < 8; ++v) {
                // B 32x16: lane<16: K {0..7,16..23}; lane>=16: {8..15,24..31}
                const int kk = 2 * (v & 3) + laneHi * 8 + (v >> 2) * 16;
                b[j].u[v] = *reinterpret_cast<const unsigned int*>(bp + kk);
            }
        }

        // --- 8 WMMAs per K-step: A fragment reused across 4 N tiles --------
#pragma unroll
        for (int i = 0; i < 2; ++i)
#pragma unroll
            for (int j = 0; j < 4; ++j)
                acc[i][j] = __builtin_amdgcn_wmma_f32_16x16x32_f16(
                    false, a[i].v, false, b[j].v, (short)0, acc[i][j],
                    false, false);
    }

    // --- epilogue: C/D layout -> VGPR r holds M = r + 8*laneHi, N = lane16 --
#pragma unroll
    for (int i = 0; i < 2; ++i) {
#pragma unroll
        for (int j = 0; j < 4; ++j) {
            const int n = blockN + waveN + j * 16 + lane16;
            if (n >= N) continue;
#pragma unroll
            for (int r = 0; r < 8; ++r) {
                const int m = blockM + waveM + i * 16 + laneHi * 8 + r;
                const float v = acc[i][j][r];
                const size_t idx = (size_t)m * N + n;
                if (EPI == 0) {
                    outH[idx] = (_Float16)v;
                } else if (EPI == 1) {
                    const float s = v / (1.0f + __expf(-v));   // silu
                    outF[idx] = s;
                    outH[idx] = (_Float16)s;
                } else {
                    const float base = baseF[idx];
                    outF[idx] = base + ((sel[m] >= 0) ? 0.1f * v : 0.0f);
                }
            }
        }
    }
}

// ---------------------------------------------------------------------------
// Per-token winning-expert adapter (A x A) + LayerNorm, full f32.
// One 192-thread workgroup per token. Thread c computes h[c]; wave 0 reduces.
// ---------------------------------------------------------------------------
__global__ __launch_bounds__(192)
void adapter_ln(const _Float16* __restrict__ preH,
                const float* __restrict__ ew,         // [M, 8]
                const float* __restrict__ adapterW,   // [8, A, A]
                const float* __restrict__ gamma,      // [8, A]
                const float* __restrict__ beta,       // [8, A]
                _Float16* __restrict__ hlnH,          // [M, A]
                int* __restrict__ selOut) {
    constexpr int A = 192;
    const int m = blockIdx.x;
    const int c = threadIdx.x;

    __shared__ float preS[A];
    __shared__ float hS[A];
    __shared__ int   selS;
    __shared__ float stat[2];

    if (c == 0) {
        int s = -1;
        const float* w = ew + (size_t)m * 8;
#pragma unroll
        for (int i = 0; i < 8; ++i)
            if (w[i] > 0.0f) s = i;            // later experts win
        selS = s;
        selOut[m] = s;
    }
    preS[c] = (float)preH[(size_t)m * A + c];
    __syncthreads();

    const int e = (selS < 0) ? 0 : selS;       // clamp; masked out later anyway
    const float* W = adapterW + ((size_t)e * A + c) * A;
    float acc = 0.0f;
    for (int a = 0; a < A; ++a) acc += preS[a] * W[a];
    hS[c] = acc;
    __syncthreads();

    if (c < 32) {
        float s = 0.0f, s2 = 0.0f;
#pragma unroll
        for (int j = 0; j < 6; ++j) {
            const float v = hS[c + 32 * j];
            s  += v;
            s2 += v * v;
        }
#pragma unroll
        for (int off = 16; off > 0; off >>= 1) {
            s  += __shfl_xor(s,  off, 32);
            s2 += __shfl_xor(s2, off, 32);
        }
        if (c == 0) {
            const float mu  = s / (float)A;
            const float var = s2 / (float)A - mu * mu;
            stat[0] = mu;
            stat[1] = rsqrtf(var + 1e-5f);
        }
    }
    __syncthreads();

    const float v = (hS[c] - stat[0]) * stat[1] * gamma[(size_t)e * A + c]
                    + beta[(size_t)e * A + c];
    hlnH[(size_t)m * A + c] = (_Float16)v;
}

extern "C" void kernel_launch(void* const* d_in, const int* in_sizes, int n_in,
                              void* d_out, int out_size, void* d_ws, size_t ws_size,
                              hipStream_t stream) {
    const float* x        = (const float*)d_in[0];   // [2,1024,768]
    const float* ew       = (const float*)d_in[1];   // [2,1024,8]
    const float* up_w     = (const float*)d_in[2];   // [3072,768]
    const float* adapt_w  = (const float*)d_in[3];   // [192,3072]
    const float* adapterW = (const float*)d_in[4];   // [8,192,192]
    const float* gamma    = (const float*)d_in[5];   // [8,192]
    const float* beta     = (const float*)d_in[6];   // [8,192]
    const float* eproj    = (const float*)d_in[7];   // [3072,192]
    const float* oproj    = (const float*)d_in[8];   // [3072,3072]
    float* out = (float*)d_out;                      // [2,1024,3072]

    const int M = 2048, E = 768, H = 3072, A = 192;

    // workspace carve-out (~81 MB)
    char* ws = (char*)d_ws;
    auto carve = [&](size_t bytes) {
        char* p = ws;
        ws += (bytes + 255) & ~(size_t)255;
        return p;
    };
    _Float16* x_h      = (_Float16*)carve((size_t)M * E * 2);
    _Float16* upw_h    = (_Float16*)carve((size_t)H * E * 2);
    _Float16* adaptw_h = (_Float16*)carve((size_t)A * H * 2);
    _Float16* eproj_h  = (_Float16*)carve((size_t)H * A * 2);
    _Float16* oproj_h  = (_Float16*)carve((size_t)H * H * 2);
    float*    shared_f = (float*)   carve((size_t)M * H * 4);
    _Float16* shared_h = (_Float16*)carve((size_t)M * H * 2);
    _Float16* pre_h    = (_Float16*)carve((size_t)M * A * 2);
    _Float16* hln_h    = (_Float16*)carve((size_t)M * A * 2);
    _Float16* t_h      = (_Float16*)carve((size_t)M * H * 2);
    int*      sel      = (int*)     carve((size_t)M * 4);

    auto cvt = [&](const float* s, _Float16* d, int n) {
        int blocks = (n + 255) / 256;
        if (blocks > 4096) blocks = 4096;
        cvt_f32_to_f16<<<blocks, 256, 0, stream>>>(s, d, n);
    };
    cvt(x,       x_h,      M * E);
    cvt(up_w,    upw_h,    H * E);
    cvt(adapt_w, adaptw_h, A * H);
    cvt(eproj,   eproj_h,  H * A);
    cvt(oproj,   oproj_h,  H * H);

    const dim3 blk(256);
    // shared = silu(x @ up_w^T)
    gemm_wmma<1><<<dim3(H / 256, M / 64), blk, 0, stream>>>(
        x_h, upw_h, M, H, E, shared_f, shared_h, nullptr, nullptr);
    // pre = shared @ adapt_w^T
    gemm_wmma<0><<<dim3((A + 255) / 256, M / 64), blk, 0, stream>>>(
        shared_h, adaptw_h, M, A, H, nullptr, pre_h, nullptr, nullptr);
    // per-token winning-expert adapter + LayerNorm
    adapter_ln<<<M, 192, 0, stream>>>(pre_h, ew, adapterW, gamma, beta,
                                      hln_h, sel);
    // t = h_ln @ expert_proj^T
    gemm_wmma<0><<<dim3(H / 256, M / 64), blk, 0, stream>>>(
        hln_h, eproj_h, M, H, A, nullptr, t_h, nullptr, nullptr);
    // out = shared + 0.1 * mask * (t @ output_proj^T)
    gemm_wmma<2><<<dim3(H / 256, M / 64), blk, 0, stream>>>(
        t_h, oproj_h, M, H, H, out, nullptr, shared_f, sel);
}
